// FlowAssembly_61048665145728
// MI455X (gfx1250) — compile-verified, matrix operand order checked by reading, not run on hardware
//
#include <hip/hip_runtime.h>
#include <hip/hip_bf16.h>

typedef __attribute__((ext_vector_type(16))) _Float16 v16h;
typedef __attribute__((ext_vector_type(8)))  _Float16 v8h;
typedef __attribute__((ext_vector_type(8)))  float    v8f;

#define NPOINTS 8192
#define NBATCH  16
#define PTSB    8          // points per block
#define CLAMPC  1.9f
#define TWOPI_I 0.63661977236758f  // 2/pi

// Combine two contiguous 8-half LDS chunks into one WMMA 16-half operand.
__device__ __forceinline__ v16h ld16(const _Float16* a, const _Float16* b) {
  v8h lo = *(const v8h*)a;
  v8h hi = *(const v8h*)b;
  return __builtin_shufflevector(lo, hi, 0,1,2,3,4,5,6,7,8,9,10,11,12,13,14,15);
}

__device__ __forceinline__ v8f wmma16(v16h a, v16h b, v8f c) {
  return __builtin_amdgcn_wmma_f32_16x16x32_f16(false, a, false, b, (short)0, c,
                                                false, false);
}

// ---------------------------------------------------------------------------
// Kernel 1: z = reverse(x)*exp(an_logs)+an_bias ; z1 (fp16) to workspace;
//           init log_det[b] = sum(an_logs)*N.
// ---------------------------------------------------------------------------
__global__ void prep_kernel(const float* __restrict__ x,
                            const float* __restrict__ logs,
                            const float* __restrict__ bias,
                            float* __restrict__ zmat,
                            float* __restrict__ logdet,
                            _Float16* __restrict__ z1h, int total) {
  int p = blockIdx.x * blockDim.x + threadIdx.x;
  if (p < total) {
    const float* xr = x + (size_t)p * 32;
    float* zr = zmat + (size_t)p * 32;
    _Float16* hr = z1h + (size_t)p * 16;
#pragma unroll
    for (int c = 0; c < 32; ++c) {
      float zc = xr[31 - c] * expf(logs[c]) + bias[c];
      zr[c] = zc;
      if (c < 16) hr[c] = (_Float16)zc;
    }
  }
  if (p < NBATCH) {
    float su = 0.f;
#pragma unroll
    for (int c = 0; c < 32; ++c) su += logs[c];
    logdet[p] = su * (float)NPOINTS;
  }
}

// ---------------------------------------------------------------------------
// Kernel 2: one coupling layer. Weight pointers pre-offset to set pair {u,u+1}.
// set 0 = s-branch (soft-clamped), set 1 = t-branch.
// ---------------------------------------------------------------------------
__global__ __launch_bounds__(128) void conv_kernel(
    const _Float16* __restrict__ z1h, const int* __restrict__ knn,
    const float* __restrict__ W1g, const float* __restrict__ b1g,
    const float* __restrict__ W2g, const float* __restrict__ b2g,
    const float* __restrict__ W3g, const float* __restrict__ b3g,
    float* __restrict__ zmat, float* __restrict__ logdet) {
  // LDS staging. Rows padded to 16B-aligned strides (40 / 72 halves) for
  // conflict-reduced ds_load_b128 operand fetches.
  __shared__ __align__(16) _Float16 sW1[2][64][40];  // [set][jout][k(32)] folded [nb|ctr]
  __shared__ __align__(16) _Float16 sW2[2][64][72];  // [set][jout][kin(64)]
  __shared__ __align__(16) _Float16 sW3[2][16][72];  // [set][jout(16)][kin(64)]
  __shared__ __align__(16) _Float16 sA[128][40];     // edge rows [nb(16)|ctr(16)]
  __shared__ __align__(16) _Float16 sY[4][16][72];   // per-wave relu(y1)
  __shared__ __align__(16) _Float16 sZ[4][16][72];   // per-wave relu(y2)
  __shared__ float sB1[2][64], sB2[2][64], sB3[2][16];
  __shared__ float sLog;

  const int tid  = threadIdx.x;
  const int wid  = tid >> 5;
  const int lane = tid & 31;
  const int n16  = lane & 15;
  const int hl   = lane >> 4;
  const int base0 = hl * 8;

  const int bi = blockIdx.x / (NPOINTS / PTSB);
  const int pb = (blockIdx.x % (NPOINTS / PTSB)) * PTSB;

  // ---- stage weights: fold W1 (48x64 -> 32x64) and transpose to [jout][k]
  for (int idx = tid; idx < 2 * 64 * 32; idx += 128) {
    int s = idx >> 11, r = idx & 2047;
    int j = r >> 5, k = r & 31;
    float v;
    if (k < 16)  // K 0..15 multiplies nb:  W1[16+k] + W1[32+k]
      v = W1g[(s * 48 + 16 + k) * 64 + j] + W1g[(s * 48 + 32 + k) * 64 + j];
    else {       // K 16..31 multiplies ctr: W1[k-16] - W1[32+k-16]
      int kk = k - 16;
      v = W1g[(s * 48 + kk) * 64 + j] - W1g[(s * 48 + 32 + kk) * 64 + j];
    }
    sW1[s][j][k] = (_Float16)v;
  }
  for (int idx = tid; idx < 2 * 64 * 64; idx += 128) {
    int s = idx >> 12, r = idx & 4095;
    int j = r >> 6, k = r & 63;
    sW2[s][j][k] = (_Float16)W2g[(s * 64 + k) * 64 + j];
  }
  for (int idx = tid; idx < 2 * 16 * 64; idx += 128) {
    int s = idx >> 10, r = idx & 1023;
    int j = r >> 6, k = r & 63;
    sW3[s][j][k] = (_Float16)W3g[(s * 64 + k) * 16 + j];
  }
  {
    int s = tid >> 6, j = tid & 63;
    sB1[s][j] = b1g[s * 64 + j];
    sB2[s][j] = b2g[s * 64 + j];
    if (tid < 32) sB3[tid >> 4][tid & 15] = b3g[(tid >> 4) * 16 + (tid & 15)];
    if (tid == 0) sLog = 0.f;
  }

  // ---- gather edge rows: each thread = one edge
  {
    const int e = tid, lp = e >> 4, k = e & 15;
    const size_t gp = (size_t)bi * NPOINTS + pb + lp;
    const int nidx = knn[gp * 16 + k];
    const _Float16* nb = z1h + ((size_t)bi * NPOINTS + nidx) * 16;
    const _Float16* ct = z1h + gp * 16;
    *(v8h*)&sA[e][0]  = *(const v8h*)nb;
    *(v8h*)&sA[e][8]  = *(const v8h*)(nb + 8);
    *(v8h*)&sA[e][16] = *(const v8h*)ct;
    *(v8h*)&sA[e][24] = *(const v8h*)(ct + 8);
  }
  __syncthreads();

  // ---- per-wave WMMA chains: wave handles tiles wid and wid+4 (1 tile = 1 point)
  for (int tt = 0; tt < 2; ++tt) {
    const int tile = wid + tt * 4;
    const int p = pb + tile;
    const int e0 = tile * 16;
    const int arow = e0 + n16;
    const v16h A1 = ld16(&sA[arow][base0], &sA[arow][16 + base0]);
    float sv = 0.f;

#pragma unroll
    for (int s = 0; s < 2; ++s) {
      // layer 1: (16 edges x 32) @ (32 x 64), bias in C, relu -> sY
#pragma unroll
      for (int nc = 0; nc < 4; ++nc) {
        const int j = nc * 16 + n16;
        const v16h B1 = ld16(&sW1[s][j][base0], &sW1[s][j][16 + base0]);
        v8f c; const float bv = sB1[s][j];
#pragma unroll
        for (int r = 0; r < 8; ++r) c[r] = bv;
        c = wmma16(A1, B1, c);
#pragma unroll
        for (int r = 0; r < 8; ++r)
          sY[wid][r + 8 * hl][j] = (_Float16)fmaxf(c[r], 0.0f);
      }
      __builtin_amdgcn_wave_barrier();

      // layer 2: (16 x 64) @ (64 x 64) as two K32 WMMAs per N-chunk -> sZ
      const v16h A2a = ld16(&sY[wid][n16][base0],      &sY[wid][n16][16 + base0]);
      const v16h A2b = ld16(&sY[wid][n16][32 + base0], &sY[wid][n16][48 + base0]);
#pragma unroll
      for (int nc = 0; nc < 4; ++nc) {
        const int j = nc * 16 + n16;
        const v16h B2a = ld16(&sW2[s][j][base0],      &sW2[s][j][16 + base0]);
        const v16h B2b = ld16(&sW2[s][j][32 + base0], &sW2[s][j][48 + base0]);
        v8f c; const float bv = sB2[s][j];
#pragma unroll
        for (int r = 0; r < 8; ++r) c[r] = bv;
        c = wmma16(A2a, B2a, c);
        c = wmma16(A2b, B2b, c);
#pragma unroll
        for (int r = 0; r < 8; ++r)
          sZ[wid][r + 8 * hl][j] = (_Float16)fmaxf(c[r], 0.0f);
      }
      __builtin_amdgcn_wave_barrier();

      // layer 3: (16 x 64) @ (64 x 16), then max over the K=16 rows
      const v16h A3a = ld16(&sZ[wid][n16][base0],      &sZ[wid][n16][16 + base0]);
      const v16h A3b = ld16(&sZ[wid][n16][32 + base0], &sZ[wid][n16][48 + base0]);
      const v16h B3a = ld16(&sW3[s][n16][base0],       &sW3[s][n16][16 + base0]);
      const v16h B3b = ld16(&sW3[s][n16][32 + base0],  &sW3[s][n16][48 + base0]);
      v8f c; const float bv = sB3[s][n16];
#pragma unroll
      for (int r = 0; r < 8; ++r) c[r] = bv;
      c = wmma16(A3a, B3a, c);
      c = wmma16(A3b, B3b, c);
      float v = c[0];
#pragma unroll
      for (int r = 1; r < 8; ++r) v = fmaxf(v, c[r]);
      v = fmaxf(v, __shfl_xor(v, 16, 32));   // combine M halves

      if (s == 0) {
        sv = CLAMPC * TWOPI_I * atanf(v * (1.0f / CLAMPC));  // soft clamp
        float r0 = sv;                                        // sum over 16 feats
        r0 += __shfl_xor(r0, 1, 32);
        r0 += __shfl_xor(r0, 2, 32);
        r0 += __shfl_xor(r0, 4, 32);
        r0 += __shfl_xor(r0, 8, 32);
        if (lane == 0) atomicAdd(&sLog, r0);
      } else {
        if (lane < 16) {
          float* zp = zmat + ((size_t)bi * NPOINTS + p) * 32 + 16 + n16;
          *zp = *zp * __expf(sv) + v;        // z2 = z2*exp(s) + t
        }
      }
      __builtin_amdgcn_wave_barrier();
    }
  }
  __syncthreads();
  if (tid == 0) atomicAdd(&logdet[bi], sLog);
}

// ---------------------------------------------------------------------------
extern "C" void kernel_launch(void* const* d_in, const int* in_sizes, int n_in,
                              void* d_out, int out_size, void* d_ws, size_t ws_size,
                              hipStream_t stream) {
  const float* x     = (const float*)d_in[0];
  const int*   knn   = (const int*)d_in[1];
  const float* logs  = (const float*)d_in[2];
  const float* biasv = (const float*)d_in[3];
  const float* W1    = (const float*)d_in[4];
  const float* b1    = (const float*)d_in[5];
  const float* W2    = (const float*)d_in[6];
  const float* b2    = (const float*)d_in[7];
  const float* W3    = (const float*)d_in[8];
  const float* b3    = (const float*)d_in[9];

  float* zmat   = (float*)d_out;
  float* logdet = zmat + (size_t)NBATCH * NPOINTS * 32;
  _Float16* z1h = (_Float16*)d_ws;   // NBATCH*NPOINTS*16 halves = 4 MB

  const int total = NBATCH * NPOINTS;
  prep_kernel<<<(total + 255) / 256, 256, 0, stream>>>(x, logs, biasv, zmat,
                                                       logdet, z1h, total);

  const dim3 grid(NBATCH * (NPOINTS / PTSB));
  for (int u = 0; u < 4; u += 2) {
    conv_kernel<<<grid, 128, 0, stream>>>(
        z1h, knn,
        W1 + (size_t)u * 48 * 64, b1 + (size_t)u * 64,
        W2 + (size_t)u * 64 * 64, b2 + (size_t)u * 64,
        W3 + (size_t)u * 64 * 16, b3 + (size_t)u * 16,
        zmat, logdet);
  }
}
